// GATv2_89764816486784
// MI455X (gfx1250) — compile-verified, matrix operand order checked by reading, not run on hardware
//
#include <hip/hip_runtime.h>
#include <hip/hip_fp16.h>
#include <cstdint>
#include <cstddef>

// ---------------------------------------------------------------------------
// GATv2Conv + pool + MLP head for MI455X (gfx1250, wave32, WMMA)
// ---------------------------------------------------------------------------

typedef __attribute__((ext_vector_type(16))) _Float16 v16h;
typedef __attribute__((ext_vector_type(8)))  float    v8f;
typedef __attribute__((ext_vector_type(4)))  _Float16 h4;

#define N_NODES   20000
#define E_EDGES   320000
#define IN_F      128
#define H_HEADS   8
#define C_CH      128
#define HCF       1024          // H_HEADS * C_CH
#define G_GRAPHS  64
#define NEG_SLOPE 0.2f

// ---- float atomics ---------------------------------------------------------

__device__ inline void atomAddF(float* p, float v) {
  // lowers to global_atomic_add_f32 (AGG/DEN/POOL are coarse-grained HBM)
  unsafeAtomicAdd(p, v);
}

__device__ inline void atomMaxF(float* addr, float val) {
  // classic IEEE-ordered-bits trick; storage always holds raw float bits.
  if (val >= 0.0f)
    atomicMax((int*)addr, __float_as_int(val));
  else
    atomicMin((unsigned int*)addr, __float_as_uint(val));
}

// ---- init: MROW=-inf, DEN=0, POOL=0, CNT=0 --------------------------------

__global__ void k_init(float* __restrict__ MROW, float* __restrict__ DEN,
                       float* __restrict__ POOL, float* __restrict__ CNT) {
  int t = blockIdx.x * blockDim.x + threadIdx.x;
  const int nh = N_NODES * H_HEADS;
  if (t < nh) {
    ((int*)MROW)[t] = 0xFF800000;   // -inf
    DEN[t] = 0.0f;
  }
  if (t < G_GRAPHS * C_CH) POOL[t] = 0.0f;
  if (t < G_GRAPHS)        CNT[t]  = 0.0f;
}

// ---- fused node transforms: XLh = f16(x@Wl+bl), XRh = f16(x@Wr+br),
//      AGG = x@Wres + bias_attn  (residual pre-seeded into accumulator) -----

__global__ __launch_bounds__(256) void k_transform(
    const float* __restrict__ x,
    const float* __restrict__ Wl,   const float* __restrict__ bl,
    const float* __restrict__ Wr,   const float* __restrict__ br,
    const float* __restrict__ Wres, const float* __restrict__ bias_attn,
    _Float16* __restrict__ XLh, _Float16* __restrict__ XRh,
    float* __restrict__ AGG)
{
  __shared__ _Float16 sx[16 * 136];          // 16 rows x 128 K, padded stride
  const int rb  = blockIdx.x * 16;
  const int tid = threadIdx.x;

  // stage x tile (f32 -> f16) into LDS
  for (int i = tid; i < 16 * IN_F; i += 256) {
    int r = i >> 7, k = i & 127;
    sx[r * 136 + k] = (_Float16)x[(size_t)(rb + r) * IN_F + k];
  }
  __syncthreads();

  const int wid  = tid >> 5;
  const int lane = tid & 31;
  const int m    = lane & 15;
  const int hlf  = lane >> 4;

  // build the four 16x32 A fragments (ISA 16-bit A layout: lanes 0-15 hold
  // K = {0..7,16..23}, lanes 16-31 hold K = {8..15,24..31} per 32-K step)
  v16h afrag[4];
  #pragma unroll
  for (int kb = 0; kb < 4; ++kb) {
    #pragma unroll
    for (int jb = 0; jb < 2; ++jb) {
      const int ks = kb * 32 + jb * 16 + hlf * 8;
      h4 lo = *(const h4*)&sx[m * 136 + ks];
      h4 hi = *(const h4*)&sx[m * 136 + ks + 4];
      #pragma unroll
      for (int t = 0; t < 4; ++t) {
        afrag[kb][jb * 8 + t]     = lo[t];
        afrag[kb][jb * 8 + 4 + t] = hi[t];
      }
    }
  }

  // 3 weight matrices x 64 column tiles = 192 tiles; 8 waves -> 24 each.
  // 192 % 8 == 0 so all waves stay fully converged at every WMMA.
  for (int t = wid; t < 3 * (HCF / 16); t += 8) {
    const int w       = t >> 6;          // 0=Wl 1=Wr 2=Wres
    const int colBase = (t & 63) * 16;
    const int col     = colBase + m;     // this lane's N index

    const float* W  = (w == 0) ? Wl : (w == 1) ? Wr : Wres;
    const float* bv = (w == 0) ? bl : (w == 1) ? br : bias_attn;

    v8f acc;
    const float b0 = bv[col];
    #pragma unroll
    for (int r = 0; r < 8; ++r) acc[r] = b0;   // bias folded into C

    #pragma unroll
    for (int kb = 0; kb < 4; ++kb) {
      v16h bfrag;
      #pragma unroll
      for (int j = 0; j < 16; ++j) {
        const int k = kb * 32 + (j >> 3) * 16 + hlf * 8 + (j & 7);
        bfrag[j] = (_Float16)W[(size_t)k * HCF + col];
      }
      acc = __builtin_amdgcn_wmma_f32_16x16x32_f16(
          false, afrag[kb], false, bfrag, (short)0, acc, false, false);
    }

    // D layout: VGPR r -> M = hlf*8 + r, N = lane&15
    if (w == 0) {
      #pragma unroll
      for (int r = 0; r < 8; ++r)
        XLh[(size_t)(rb + hlf * 8 + r) * HCF + col] = (_Float16)acc[r];
    } else if (w == 1) {
      #pragma unroll
      for (int r = 0; r < 8; ++r)
        XRh[(size_t)(rb + hlf * 8 + r) * HCF + col] = (_Float16)acc[r];
    } else {
      #pragma unroll
      for (int r = 0; r < 8; ++r)
        AGG[(size_t)(rb + hlf * 8 + r) * HCF + col] = acc[r];
    }
  }
}

// ---- per-edge attention logits + segment max ------------------------------
// one block per edge; one wave per head; lane covers 4 channels.

__global__ __launch_bounds__(256) void k_edge_logits(
    const _Float16* __restrict__ XLh, const _Float16* __restrict__ XRh,
    const int* __restrict__ src, const int* __restrict__ dst,
    const float* __restrict__ att,
    float* __restrict__ ELOG, float* __restrict__ MROW)
{
  const int e    = blockIdx.x;
  const int h    = threadIdx.x >> 5;
  const int lane = threadIdx.x & 31;
  const int s = src[e], d = dst[e];

  const size_t boff = (size_t)h * C_CH + (size_t)lane * 4;
  h4 xl = *(const h4*)&XLh[(size_t)s * HCF + boff];
  h4 xr = *(const h4*)&XRh[(size_t)d * HCF + boff];
  const float4 a = *(const float4*)&att[h * C_CH + lane * 4];
  const float av[4] = {a.x, a.y, a.z, a.w};

  float sum = 0.0f;
  #pragma unroll
  for (int i = 0; i < 4; ++i) {
    float v = (float)xl[i] + (float)xr[i];
    v = (v > 0.0f) ? v : NEG_SLOPE * v;
    sum += av[i] * v;
  }
  #pragma unroll
  for (int off = 16; off >= 1; off >>= 1)
    sum += __shfl_xor(sum, off, 32);

  if (lane == 0) {
    ELOG[(size_t)e * H_HEADS + h] = sum;
    atomMaxF(&MROW[(size_t)d * H_HEADS + h], sum);
  }
}

// ---- exp(e - m[dst]) in place + segment denominator -----------------------

__global__ void k_expsum(const int* __restrict__ dst,
                         float* __restrict__ ELOG,
                         const float* __restrict__ MROW,
                         float* __restrict__ DEN)
{
  int t = blockIdx.x * blockDim.x + threadIdx.x;
  if (t >= E_EDGES * H_HEADS) return;
  const int e = t >> 3, h = t & 7;
  const int d = dst[e];
  const float v = expf(ELOG[t] - MROW[d * H_HEADS + h]);
  ELOG[t] = v;
  atomAddF(&DEN[d * H_HEADS + h], v);
}

// ---- alpha-weighted scatter aggregation: AGG[dst] += alpha * xl[src] ------

__global__ __launch_bounds__(256) void k_aggregate(
    const _Float16* __restrict__ XLh,
    const int* __restrict__ src, const int* __restrict__ dst,
    const float* __restrict__ ELOG, const float* __restrict__ DEN,
    float* __restrict__ AGG)
{
  const int e    = blockIdx.x;
  const int h    = threadIdx.x >> 5;
  const int lane = threadIdx.x & 31;
  const int s = src[e], d = dst[e];
  const float w = ELOG[(size_t)e * H_HEADS + h] / DEN[(size_t)d * H_HEADS + h];
  const size_t sbase = (size_t)s * HCF + (size_t)h * C_CH;
  const size_t dbase = (size_t)d * HCF + (size_t)h * C_CH;
  #pragma unroll
  for (int i = 0; i < 4; ++i) {
    const int c = lane + 32 * i;
    atomAddF(&AGG[dbase + c], w * (float)XLh[sbase + c]);
  }
}

// ---- post-attention Linear(1024->128) + ELU via WMMA ----------------------

__global__ __launch_bounds__(256) void k_post(
    const float* __restrict__ AGG,
    const float* __restrict__ W_lin, const float* __restrict__ b_lin,
    float* __restrict__ HB)
{
  __shared__ _Float16 sa[16 * 1032];        // 16 x 1024, padded stride
  const int rb  = blockIdx.x * 16;
  const int tid = threadIdx.x;

  for (int i = tid; i < 16 * HCF; i += 256) {
    int r = i >> 10, k = i & 1023;
    sa[r * 1032 + k] = (_Float16)AGG[(size_t)(rb + r) * HCF + k];
  }
  __syncthreads();

  const int wid  = tid >> 5;
  const int lane = tid & 31;
  const int m    = lane & 15;
  const int hlf  = lane >> 4;
  const int col  = wid * 16 + m;     // 8 waves cover the 128 output columns

  v8f acc;
  const float b0 = b_lin[col];
  #pragma unroll
  for (int r = 0; r < 8; ++r) acc[r] = b0;

  for (int kb = 0; kb < 32; ++kb) {
    v16h afrag, bfrag;
    #pragma unroll
    for (int jb = 0; jb < 2; ++jb) {
      const int ks = kb * 32 + jb * 16 + hlf * 8;
      h4 lo = *(const h4*)&sa[m * 1032 + ks];
      h4 hi = *(const h4*)&sa[m * 1032 + ks + 4];
      #pragma unroll
      for (int t = 0; t < 4; ++t) {
        afrag[jb * 8 + t]     = lo[t];
        afrag[jb * 8 + 4 + t] = hi[t];
      }
    }
    #pragma unroll
    for (int j = 0; j < 16; ++j) {
      const int k = kb * 32 + (j >> 3) * 16 + hlf * 8 + (j & 7);
      bfrag[j] = (_Float16)W_lin[(size_t)k * C_CH + col];
    }
    acc = __builtin_amdgcn_wmma_f32_16x16x32_f16(
        false, afrag, false, bfrag, (short)0, acc, false, false);
  }

  #pragma unroll
  for (int r = 0; r < 8; ++r) {
    float v = acc[r];
    v = (v > 0.0f) ? v : (expf(v) - 1.0f);          // ELU
    HB[(size_t)(rb + hlf * 8 + r) * C_CH + col] = v;
  }
}

// ---- global mean pool (sums + counts) -------------------------------------

__global__ void k_pool(const float* __restrict__ HB,
                       const int* __restrict__ batch,
                       float* __restrict__ POOL, float* __restrict__ CNT)
{
  int t = blockIdx.x * blockDim.x + threadIdx.x;
  if (t >= N_NODES * C_CH) return;
  const int n = t >> 7, c = t & 127;
  const int b = batch[n];
  atomAddF(&POOL[b * C_CH + c], HB[t]);
  if (c == 0) atomAddF(&CNT[b], 1.0f);
}

// ---- tiny MLP head: 128 -> 16 -> 32 -> 5 (one thread per graph) -----------

__global__ void k_mlp(const float* __restrict__ POOL, const float* __restrict__ CNT,
                      const float* __restrict__ W1, const float* __restrict__ b1,
                      const float* __restrict__ W2, const float* __restrict__ b2,
                      const float* __restrict__ W3, const float* __restrict__ b3,
                      float* __restrict__ out)
{
  const int g = threadIdx.x;
  if (g >= G_GRAPHS) return;
  const float inv = 1.0f / fmaxf(CNT[g], 1.0f);

  float h1[16];
  for (int o = 0; o < 16; ++o) {
    float s = b1[o];
    for (int k = 0; k < C_CH; ++k)
      s += (POOL[g * C_CH + k] * inv) * W1[k * 16 + o];
    h1[o] = fmaxf(s, 0.0f);
  }
  float h2[32];
  for (int o = 0; o < 32; ++o) {
    float s = b2[o];
    for (int k = 0; k < 16; ++k) s += h1[k] * W2[k * 32 + o];
    h2[o] = fmaxf(s, 0.0f);
  }
  for (int o = 0; o < 5; ++o) {
    float s = b3[o];
    for (int k = 0; k < 32; ++k) s += h2[k] * W3[k * 5 + o];
    out[g * 5 + o] = s;
  }
}

// ---------------------------------------------------------------------------

extern "C" void kernel_launch(void* const* d_in, const int* in_sizes, int n_in,
                              void* d_out, int out_size, void* d_ws, size_t ws_size,
                              hipStream_t stream) {
  const float* x         = (const float*)d_in[0];
  const int*   ei        = (const int*)d_in[1];       // [2,E] flat: src then dst
  const int*   batch     = (const int*)d_in[2];
  const float* Wl        = (const float*)d_in[3];
  const float* bl        = (const float*)d_in[4];
  const float* Wr        = (const float*)d_in[5];
  const float* br        = (const float*)d_in[6];
  const float* att       = (const float*)d_in[7];
  const float* Wres      = (const float*)d_in[8];
  const float* bias_attn = (const float*)d_in[9];
  const float* W_lin     = (const float*)d_in[10];
  const float* b_lin     = (const float*)d_in[11];
  const float* W1 = (const float*)d_in[12];
  const float* b1 = (const float*)d_in[13];
  const float* W2 = (const float*)d_in[14];
  const float* b2 = (const float*)d_in[15];
  const float* W3 = (const float*)d_in[16];
  const float* b3 = (const float*)d_in[17];
  float* out = (float*)d_out;

  const int* src = ei;
  const int* dst = ei + E_EDGES;

  // workspace carve-out (256B aligned): ~186 MB total
  char* p = (char*)d_ws;
  auto alloc = [&](size_t bytes) -> char* {
    char* q = p;
    p += (bytes + 255) & ~(size_t)255;
    return q;
  };
  _Float16* XLh  = (_Float16*)alloc((size_t)N_NODES * HCF * 2);
  _Float16* XRh  = (_Float16*)alloc((size_t)N_NODES * HCF * 2);
  float*    AGG  = (float*)   alloc((size_t)N_NODES * HCF * 4);
  float*    ELOG = (float*)   alloc((size_t)E_EDGES * H_HEADS * 4);
  float*    MROW = (float*)   alloc((size_t)N_NODES * H_HEADS * 4);
  float*    DEN  = (float*)   alloc((size_t)N_NODES * H_HEADS * 4);
  float*    HB   = (float*)   alloc((size_t)N_NODES * C_CH * 4);
  float*    POOL = (float*)   alloc((size_t)G_GRAPHS * C_CH * 4);
  float*    CNT  = (float*)   alloc((size_t)G_GRAPHS * 4);

  const int initN = N_NODES * H_HEADS;              // largest init range
  k_init<<<(initN + 255) / 256, 256, 0, stream>>>(MROW, DEN, POOL, CNT);

  k_transform<<<N_NODES / 16, 256, 0, stream>>>(
      x, Wl, bl, Wr, br, Wres, bias_attn, XLh, XRh, AGG);

  k_edge_logits<<<E_EDGES, 256, 0, stream>>>(XLh, XRh, src, dst, att, ELOG, MROW);

  k_expsum<<<(E_EDGES * H_HEADS + 255) / 256, 256, 0, stream>>>(dst, ELOG, MROW, DEN);

  k_aggregate<<<E_EDGES, 256, 0, stream>>>(XLh, src, dst, ELOG, DEN, AGG);

  k_post<<<N_NODES / 16, 256, 0, stream>>>(AGG, W_lin, b_lin, HB);

  k_pool<<<(N_NODES * C_CH + 255) / 256, 256, 0, stream>>>(HB, batch, POOL, CNT);

  k_mlp<<<1, 64, 0, stream>>>(POOL, CNT, W1, b1, W2, b2, W3, b3, out);
}
